// InvariantTransformerMessage_13005160972669
// MI455X (gfx1250) — matrix operand order repaired and checked by reading, not compiled, not run on hardware
//
#include <hip/hip_runtime.h>
#include <hip/hip_bf16.h>
#include <cstdint>

// ---------------- CDNA5 WMMA types ----------------
typedef __attribute__((ext_vector_type(16))) __bf16 v16bf;
typedef __attribute__((ext_vector_type(8)))  float  v8f;

union Frag {
    uint32_t u[8];
    uint4    q[2];
    v16bf    v;
};

#define FDIM 128
#define HFD  512
#define ODIM 384
#define PI_F 3.14159265358979f
#define CUTOFF_F 5.0f

// ---------------- helpers ----------------
__device__ __forceinline__ uint32_t f2bf1(float x) {
    uint32_t u = __float_as_uint(x);
    return (u + 0x7FFFu + ((u >> 16) & 1u)) >> 16;
}
__device__ __forceinline__ uint32_t packbf(float lo, float hi) {
    return f2bf1(lo) | (f2bf1(hi) << 16);
}
__device__ __forceinline__ float bf2f(unsigned short b) {
    return __uint_as_float(((uint32_t)b) << 16);
}
__device__ __forceinline__ float silu(float x) {
    return x / (1.f + __expf(-x));
}

// ---------------------------------------------------------------------------
// Pack a row-major f32 weight W[K x N] into bf16 WMMA B-fragment order.
// Fragment (kt, nt) occupies 256 dwords; dword index = lane*8 + r so the
// consumer wave loads 32B contiguous per lane (two global_load_b128).
// B layout (16x16x32 bf16): K = 2r + 8*(r>=4) + 8*(lane>=16), pairs K,K+1 (matches A).
// Rows >= Kvalid are zero (K padding for the RBF GEMM, 20 -> 32).
// ---------------------------------------------------------------------------
__global__ void pack_b_kernel(const float* __restrict__ W, uint32_t* __restrict__ out,
                              int Kvalid, int N, int nKt) {
    int g = blockIdx.x * blockDim.x + threadIdx.x;
    int nNt = N >> 4;
    int total = nKt * nNt * 256;
    if (g >= total) return;
    int frag = g >> 8, slot = g & 255;
    int lane = slot >> 3, r = slot & 7;
    int kt = frag / nNt, nt = frag % nNt;
    int n  = nt * 16 + (lane & 15);
    int K0 = 2 * r + ((r >> 2) << 3) + ((lane >> 4) << 3);
    int k0 = kt * 32 + K0;
    float lo = (k0     < Kvalid) ? W[(size_t)k0 * N + n]       : 0.f;
    float hi = (k0 + 1 < Kvalid) ? W[(size_t)(k0 + 1) * N + n] : 0.f;
    out[g] = packbf(lo, hi);
}

// ---------------------------------------------------------------------------
// Kernel 1: LayerNorm + q/k/v projections. 32 nodes per block, 8 waves.
// ---------------------------------------------------------------------------
__global__ __launch_bounds__(256) void qkv_kernel(
    const float* __restrict__ s, const float* __restrict__ ln_g, const float* __restrict__ ln_b,
    const uint32_t* __restrict__ Wqp, const uint32_t* __restrict__ Wkp, const uint32_t* __restrict__ Wvp,
    const float* __restrict__ bq, const float* __restrict__ bk, const float* __restrict__ bv,
    float* __restrict__ q, float* __restrict__ k, float* __restrict__ v, int N)
{
    __shared__ float    inp_s[32][FDIM];
    __shared__ float    red[32][8][2];
    __shared__ float    stat[32][2];
    __shared__ uint32_t afrag[2][4][256];

    int t = threadIdx.x;
    int node0 = blockIdx.x * 32;
    bool full_tile = (node0 + 32 <= N);     // block-uniform guard
    int row = t >> 3, part = t & 7;
    int gn = min(node0 + row, N - 1);

    // ---- LayerNorm ----
    float xs[16];
    {
        const float4* src = (const float4*)(s + (size_t)gn * FDIM + part * 16);
        float4* xv = (float4*)xs;
        xv[0] = src[0]; xv[1] = src[1]; xv[2] = src[2]; xv[3] = src[3];
    }
    float sm = 0.f, sq = 0.f;
#pragma unroll
    for (int e = 0; e < 16; e++) { sm += xs[e]; sq += xs[e] * xs[e]; }
    red[row][part][0] = sm; red[row][part][1] = sq;
    __syncthreads();
    if (part == 0) {
        float S = 0.f, Q = 0.f;
#pragma unroll
        for (int p = 0; p < 8; p++) { S += red[row][p][0]; Q += red[row][p][1]; }
        float mu  = S * (1.f / FDIM);
        float var = Q * (1.f / FDIM) - mu * mu;
        stat[row][0] = mu;
        stat[row][1] = rsqrtf(var + 1e-5f);
    }
    __syncthreads();
    {
        float mu = stat[row][0], rs = stat[row][1];
#pragma unroll
        for (int e = 0; e < 16; e++) {
            int f = part * 16 + e;
            inp_s[row][f] = (xs[e] - mu) * rs * ln_g[f] + ln_b[f];
        }
    }
    __syncthreads();

    // ---- pack A fragments (2 M-tiles x 4 K-tiles) ----
    for (int g = t; g < 2048; g += 256) {
        int mt = g >> 10, kt = (g >> 8) & 3, slot = g & 255;
        int lane = slot >> 3, r = slot & 7;
        int M  = lane & 15;
        int K0 = 2 * r + ((r >> 2) << 3) + ((lane >> 4) << 3);
        int f  = kt * 32 + K0;
        afrag[mt][kt][slot] = packbf(inp_s[mt * 16 + M][f], inp_s[mt * 16 + M][f + 1]);
    }
    __syncthreads();

    // ---- GEMM: 3 matrices x 2 M-tiles x 32 N-tiles ----
    int wave = t >> 5, lane = t & 31;
    for (int c = wave; c < 192; c += 8) {
        int m = c / 64, rem = c % 64, mt = rem >> 5, nt = rem & 31;
        const uint32_t* wp = (m == 0) ? Wqp : (m == 1) ? Wkp : Wvp;
        const float*    bb = (m == 0) ? bq  : (m == 1) ? bk  : bv;
        float*          op = (m == 0) ? q   : (m == 1) ? k   : v;
        v8f acc = {0.f, 0.f, 0.f, 0.f, 0.f, 0.f, 0.f, 0.f};
#pragma unroll
        for (int kt = 0; kt < 4; kt++) {
            Frag A, B;
            const uint4* ap = (const uint4*)&afrag[mt][kt][lane * 8];
            A.q[0] = ap[0]; A.q[1] = ap[1];
            const uint4* bp = (const uint4*)(wp + ((size_t)(kt * 32 + nt) * 256 + lane * 8));
            B.q[0] = bp[0]; B.q[1] = bp[1];
            acc = __builtin_amdgcn_wmma_f32_16x16x32_bf16(false, A.v, false, B.v,
                                                          (short)0, acc, false, false);
        }
        int col   = nt * 16 + (lane & 15);
        float bia = bb[col];
        int rbase = mt * 16 + ((lane >> 4) << 3);
        float* base = op + (size_t)(node0 + rbase) * HFD + col;
        if (full_tile) {                 // uniform fast path: 8 stores, imm offsets
#pragma unroll
            for (int r = 0; r < 8; r++) base[(size_t)r * HFD] = acc[r] + bia;
        } else {
#pragma unroll
            for (int r = 0; r < 8; r++)
                if (node0 + rbase + r < N) base[(size_t)r * HFD] = acc[r] + bia;
        }
    }
}

// ---------------------------------------------------------------------------
// Kernel 2: fully fused per-edge pipeline. 32 edges per block, 8 waves.
// ---------------------------------------------------------------------------
__global__ __launch_bounds__(256) void edge_kernel(
    const float* __restrict__ dist, const int* __restrict__ nbrs,
    const float* __restrict__ q, const float* __restrict__ k, const float* __restrict__ v,
    const uint32_t* __restrict__ dkWp, const float* __restrict__ dkb,
    const uint32_t* __restrict__ dvWp, const float* __restrict__ dvb,
    const uint32_t* __restrict__ denseWp, const float* __restrict__ denseb,
    float* __restrict__ out, int E, int N)
{
    __shared__ uint32_t       rfrag[2][256];
    __shared__ unsigned short dk_s[32][HFD];
    __shared__ unsigned short dv_s[32][HFD];
    __shared__ unsigned short msg_s[32][HFD];
    __shared__ float          attn_s[32][4];
    __shared__ int            ij_s[32][2];
    __shared__ float          env_s[32];

    int t = threadIdx.x;
    int e0 = blockIdx.x * 32;
    bool full_tile = (e0 + 32 <= E);        // block-uniform guard

    if (t < 32) {
        int e = min(e0 + t, E - 1);
        float d = dist[e];
        env_s[t]   = 0.5f * (__cosf(PI_F * d / CUTOFF_F) + 1.f);
        ij_s[t][0] = nbrs[2 * e];
        ij_s[t][1] = nbrs[2 * e + 1];
    }

    // ---- RBF A-fragments (K padded 20 -> 32) ----
    const float delta  = CUTOFF_F / 19.0f;
    const float rdelta = 19.0f / CUTOFF_F;
    for (int g = t; g < 512; g += 256) {
        int mt = g >> 8, slot = g & 255;
        int lane = slot >> 3, r = slot & 7;
        int M  = lane & 15;
        int K0 = 2 * r + ((r >> 2) << 3) + ((lane >> 4) << 3);
        int e  = min(e0 + mt * 16 + M, E - 1);
        float d = dist[e];
        float lo = 0.f, hi = 0.f;
        if (K0 < 20)     { float z = (d - (float)K0 * delta) * rdelta;       lo = __expf(-0.5f * z * z); }
        if (K0 + 1 < 20) { float z = (d - (float)(K0 + 1) * delta) * rdelta; hi = __expf(-0.5f * z * z); }
        rfrag[mt][slot] = packbf(lo, hi);
    }
    __syncthreads();

    int wave = t >> 5, lane = t & 31;

    // ---- dk / dv GEMM + SiLU (single wmma per C-tile since K<=32) ----
    for (int c = wave; c < 128; c += 8) {
        int m = c >> 6, mt = (c >> 5) & 1, nt = c & 31;
        const uint32_t* wp = m ? dvWp : dkWp;
        const float*    bb = m ? dvb  : dkb;
        unsigned short* ds = m ? &dv_s[0][0] : &dk_s[0][0];
        Frag A, B;
        const uint4* ap = (const uint4*)&rfrag[mt][lane * 8];
        A.q[0] = ap[0]; A.q[1] = ap[1];
        const uint4* bp = (const uint4*)(wp + ((size_t)nt * 256 + lane * 8));
        B.q[0] = bp[0]; B.q[1] = bp[1];
        v8f acc = {0.f, 0.f, 0.f, 0.f, 0.f, 0.f, 0.f, 0.f};
        acc = __builtin_amdgcn_wmma_f32_16x16x32_bf16(false, A.v, false, B.v,
                                                      (short)0, acc, false, false);
        int col   = nt * 16 + (lane & 15);
        float bia = bb[col];
        int rbase = mt * 16 + ((lane >> 4) << 3);
        unsigned short* dbase = ds + (size_t)rbase * HFD + col;
#pragma unroll
        for (int r = 0; r < 8; r++) {
            float x = silu(acc[r] + bia);
            dbase[(size_t)r * HFD] = (unsigned short)f2bf1(x);
        }
    }
    __syncthreads();

    // ---- attention: 128 (edge, head) pairs, 2 lanes each ----
    {
        int p = t >> 1, half = t & 1;
        int e = p >> 2, h = p & 3;
        int i = ij_s[e][0], j = ij_s[e][1];
        const float4* qp = (const float4*)(q + (size_t)i * HFD + h * 128 + half * 64);
        const float4* kp = (const float4*)(k + (size_t)j * HFD + h * 128 + half * 64);
        const unsigned short* dkp = &dk_s[e][h * 128 + half * 64];
        float sum = 0.f;
#pragma unroll 4
        for (int x = 0; x < 16; x++) {
            float4 qv = qp[x], kv = kp[x];
            sum += qv.x * kv.x * bf2f(dkp[4 * x + 0]);
            sum += qv.y * kv.y * bf2f(dkp[4 * x + 1]);
            sum += qv.z * kv.z * bf2f(dkp[4 * x + 2]);
            sum += qv.w * kv.w * bf2f(dkp[4 * x + 3]);
        }
        sum += __shfl_xor(sum, 1);
        if (half == 0) attn_s[e][h] = silu(sum) * env_s[e];
    }
    __syncthreads();

    // ---- msg = attn * v[j] * dv  (bf16, A-fragment source for dense GEMM) ----
    for (int x = t; x < 32 * HFD; x += 256) {
        int e = x >> 9, c = x & 511, h = c >> 7;
        int j = ij_s[e][1];
        float val = attn_s[e][h] * v[(size_t)j * HFD + c] * bf2f(dv_s[e][c]);
        msg_s[e][c] = (unsigned short)f2bf1(val);
    }
    __syncthreads();

    // ---- dense GEMM: msg(32x512) @ denseW(512x384). Wave owns one M-tile and
    //      6 N-tiles; A fragment reused across the 6 wmmas per K-step. ----
    {
        int mt  = wave >> 2;
        int ntb = wave & 3;
        v8f acc[6];
#pragma unroll
        for (int j = 0; j < 6; j++) acc[j] = (v8f){0.f, 0.f, 0.f, 0.f, 0.f, 0.f, 0.f, 0.f};
        int M = lane & 15;
        int rowA = mt * 16 + M;
#pragma unroll 2
        for (int kt = 0; kt < 16; kt++) {
            Frag A;
#pragma unroll
            for (int r = 0; r < 8; r++) {
                int K0 = 2 * r + ((r >> 2) << 3) + ((lane >> 4) << 3);
                A.u[r] = *(const uint32_t*)&msg_s[rowA][kt * 32 + K0];
            }
#pragma unroll
            for (int j = 0; j < 6; j++) {
                int nt = ntb + 4 * j;
                Frag B;
                const uint4* bp = (const uint4*)(denseWp + ((size_t)(kt * 24 + nt) * 256 + lane * 8));
                B.q[0] = bp[0]; B.q[1] = bp[1];
                acc[j] = __builtin_amdgcn_wmma_f32_16x16x32_bf16(false, A.v, false, B.v,
                                                                 (short)0, acc[j], false, false);
            }
        }
        int rbase = mt * 16 + ((lane >> 4) << 3);
        float* obase = out + (size_t)(e0 + rbase) * ODIM + ntb * 16 + (lane & 15);
        if (full_tile) {                 // uniform fast path: imm-offset stores
#pragma unroll
            for (int j = 0; j < 6; j++) {
                float bia = denseb[ntb * 16 + 64 * j + (lane & 15)];
#pragma unroll
                for (int r = 0; r < 8; r++)
                    obase[(size_t)r * ODIM + 64 * j] = acc[j][r] + bia;
            }
        } else {
#pragma unroll
            for (int j = 0; j < 6; j++) {
                float bia = denseb[ntb * 16 + 64 * j + (lane & 15)];
#pragma unroll
                for (int r = 0; r < 8; r++)
                    if (e0 + rbase + r < E) obase[(size_t)r * ODIM + 64 * j] = acc[j][r] + bia;
            }
        }
    }
}

// ---------------------------------------------------------------------------
extern "C" void kernel_launch(void* const* d_in, const int* in_sizes, int n_in,
                              void* d_out, int out_size, void* d_ws, size_t ws_size,
                              hipStream_t stream) {
    const float* s_j    = (const float*)d_in[0];
    const float* dist   = (const float*)d_in[1];
    const int*   nbrs   = (const int*)d_in[2];
    const float* ln_g   = (const float*)d_in[3];
    const float* ln_b   = (const float*)d_in[4];
    const float* Wq     = (const float*)d_in[5];
    const float* bq     = (const float*)d_in[6];
    const float* Wk     = (const float*)d_in[7];
    const float* bk     = (const float*)d_in[8];
    const float* Wv     = (const float*)d_in[9];
    const float* bv     = (const float*)d_in[10];
    const float* dkW    = (const float*)d_in[11];
    const float* dkb    = (const float*)d_in[12];
    const float* dvW    = (const float*)d_in[13];
    const float* dvb    = (const float*)d_in[14];
    const float* denseW = (const float*)d_in[15];
    const float* denseb = (const float*)d_in[16];

    int N = in_sizes[0] / FDIM;   // 20000
    int E = in_sizes[1];          // 160000
    float* out = (float*)d_out;

    // workspace layout
    char*  ws  = (char*)d_ws;
    size_t qsz = (size_t)N * HFD * sizeof(float);
    float* q = (float*)ws;
    float* k = (float*)(ws + qsz);
    float* v = (float*)(ws + 2 * qsz);
    uint32_t* Wqp  = (uint32_t*)(ws + 3 * qsz);
    uint32_t* Wkp  = Wqp  + 4  * 32 * 256;   // 128x512 -> 4 K-tiles x 32 N-tiles
    uint32_t* Wvp  = Wkp  + 4  * 32 * 256;
    uint32_t* dkWp = Wvp  + 4  * 32 * 256;
    uint32_t* dvWp = dkWp + 1  * 32 * 256;   // 32(pad)x512 -> 1 x 32
    uint32_t* dWp  = dvWp + 1  * 32 * 256;   // 512x384 -> 16 x 24

    // pre-swizzle weights to bf16 B-fragment order
    pack_b_kernel<<<(4 * 32 * 256 + 255) / 256, 256, 0, stream>>>(Wq, Wqp, 128, 512, 4);
    pack_b_kernel<<<(4 * 32 * 256 + 255) / 256, 256, 0, stream>>>(Wk, Wkp, 128, 512, 4);
    pack_b_kernel<<<(4 * 32 * 256 + 255) / 256, 256, 0, stream>>>(Wv, Wvp, 128, 512, 4);
    pack_b_kernel<<<(1 * 32 * 256 + 255) / 256, 256, 0, stream>>>(dkW, dkWp, 20, 512, 1);
    pack_b_kernel<<<(1 * 32 * 256 + 255) / 256, 256, 0, stream>>>(dvW, dvWp, 20, 512, 1);
    pack_b_kernel<<<(16 * 24 * 256 + 255) / 256, 256, 0, stream>>>(denseW, dWp, 512, 384, 16);

    // node pass: LayerNorm + q/k/v
    qkv_kernel<<<(N + 31) / 32, 256, 0, stream>>>(s_j, ln_g, ln_b, Wqp, Wkp, Wvp,
                                                  bq, bk, bv, q, k, v, N);

    // fused edge pass
    edge_kernel<<<(E + 31) / 32, 256, 0, stream>>>(dist, nbrs, q, k, v,
                                                   dkWp, dkb, dvWp, dvb,
                                                   dWp, denseb, out, E, N);
}
